// ProgramStateEncoder_53128745451858
// MI455X (gfx1250) — matrix operand order repaired and checked by reading, not compiled, not run on hardware
//
#include <hip/hip_runtime.h>

// ---------------------------------------------------------------------------
// Types for CDNA5 WMMA
// ---------------------------------------------------------------------------
typedef __attribute__((ext_vector_type(16))) __bf16 v16bfx;
typedef __attribute__((ext_vector_type(8)))  float  v8fx;

#define LEAKY_SLOPE 0.01f
#define BN_EPS 1e-5f

#define BQ   16384   // batch
#define DD   2048
#define HH1  2048
#define HH2  1024
#define PP   512
#define LL   256

// GEMM tiling
#define BM 128
#define BN 128
#define BK 32
#define LDSW 40   // ushorts per LDS row (32 data + 8 pad) = 80B, 16B-aligned

// ---------------------------------------------------------------------------
// Prep kernels (run once per launch, tiny vs GEMM cost)
// ---------------------------------------------------------------------------

// Fold BN scale into weight columns and emit TRANSPOSED bf16 weights
// Wt[N][K] from W[K][N], LDS-tiled 64x64 so both global accesses coalesce.
template <bool FOLD>
__global__ __launch_bounds__(256) void fold_transpose_kernel(
    const float* __restrict__ W, const float* __restrict__ g,
    const float* __restrict__ v, __bf16* __restrict__ Wt, int K, int N) {
    __shared__ __bf16 tilT[64][68];  // [n][k], padded rows (136B, 8B-aligned)
    __shared__ float alph[64];

    const int t  = threadIdx.x;
    const int bk = blockIdx.y * 64;
    const int bn = blockIdx.x * 64;

    if (t < 64) {
        float a_ = 1.0f;
        if constexpr (FOLD) a_ = g[bn + t] * rsqrtf(v[bn + t] + BN_EPS);
        alph[t] = a_;
    }
    __syncthreads();

#pragma unroll
    for (int r = 0; r < 4; r++) {
        int k  = r * 16 + (t >> 4);
        int n4 = (t & 15) * 4;
        float4 f = *(const float4*)(W + (size_t)(bk + k) * N + bn + n4);
        tilT[n4 + 0][k] = (__bf16)(f.x * alph[n4 + 0]);
        tilT[n4 + 1][k] = (__bf16)(f.y * alph[n4 + 1]);
        tilT[n4 + 2][k] = (__bf16)(f.z * alph[n4 + 2]);
        tilT[n4 + 3][k] = (__bf16)(f.w * alph[n4 + 3]);
    }
    __syncthreads();

#pragma unroll
    for (int w = 0; w < 4; w++) {
        int n  = w * 16 + (t >> 4);
        int k4 = (t & 15) * 4;
        uint2 d = *(const uint2*)&tilT[n][k4];
        *(uint2*)(Wt + (size_t)(bn + n) * K + bk + k4) = d;
    }
}

__global__ void fold_bias_kernel(const float* __restrict__ b,
                                 const float* __restrict__ g,
                                 const float* __restrict__ be,
                                 const float* __restrict__ m,
                                 const float* __restrict__ v,
                                 float* __restrict__ biasf, int n) {
    int i = blockIdx.x * blockDim.x + threadIdx.x;
    if (i >= n) return;
    float alpha = g[i] * rsqrtf(v[i] + BN_EPS);
    biasf[i] = (b[i] - m[i]) * alpha + be[i];
}

// One 256-thread block per basis row p: normalize to unit L2, write nb^T[L][P].
__global__ void normalize_basis_kernel(const float* __restrict__ basis,
                                       __bf16* __restrict__ nb_t) {
    __shared__ float red[LL];
    int row = blockIdx.x;
    int t = threadIdx.x;
    float v = basis[row * LL + t];
    red[t] = v * v;
    __syncthreads();
    for (int s = LL / 2; s > 0; s >>= 1) {
        if (t < s) red[t] += red[t + s];
        __syncthreads();
    }
    float scale = 1.0f / fmaxf(sqrtf(red[0]), 1e-12f);
    nb_t[(size_t)t * PP + row] = (__bf16)(v * scale);
}

// ---------------------------------------------------------------------------
// bf16 WMMA GEMM:  C = act(A @ B + bias)
//   A : [M,K] row-major, f32 (converted on LDS store) or bf16
//   Bt: [N,K] row-major bf16 (pre-transposed, BN-folded weights)
//   Block tile 128x128, K-step 32, double-buffered LDS, register staging.
//   8 waves; wave = 2x4 grid of 16x16 WMMA tiles (8 v_wmma / K-step / wave).
// ---------------------------------------------------------------------------
template <bool A_F32, bool LEAKY, bool BIAS, bool OUT_BF16>
__global__ __launch_bounds__(256) void gemm_wmma_kernel(
    const void* __restrict__ Aptr, const __bf16* __restrict__ Bt,
    const float* __restrict__ bias,
    float* __restrict__ Cf, __bf16* __restrict__ Cbf,
    int M, int N, int K) {

    __shared__ unsigned short Alds[2][BM * LDSW];  // [m][k], 80B row stride
    __shared__ unsigned short Blds[2][BN * LDSW];  // [n][k]

    const int tid   = threadIdx.x;
    const int lane  = tid & 31;
    const int wave  = tid >> 5;
    const int waveM = wave & 3;    // 4 waves along M -> 32 rows each
    const int waveN = wave >> 2;   // 2 waves along N -> 64 cols each
    const int bm = blockIdx.y * BM;
    const int bn = blockIdx.x * BN;

    const int lhalf = lane >> 4;   // K-half selector per ISA 16-bit layouts
    const int lmod  = lane & 15;   // M (A-frag) / N (B-frag, C/D) index

    // Cooperative loader indices: 32 rows x 8 col-chunks per pass, 4 passes
    const int rowL = tid >> 3;          // 0..31
    const int colL = (tid & 7) * 4;     // 0..28 step 4

    v8fx acc[2][4];
#pragma unroll
    for (int i = 0; i < 2; i++)
#pragma unroll
        for (int j = 0; j < 4; j++)
#pragma unroll
            for (int e = 0; e < 8; e++) acc[i][j][e] = 0.0f;

    float4 af4[4];
    uint2  au2[4];
    uint2  bu2[4];

    auto loadRegs = [&](int kb) {
        if constexpr (A_F32) {
            const float* A = (const float*)Aptr;
#pragma unroll
            for (int p = 0; p < 4; p++)
                af4[p] = *(const float4*)(A + (size_t)(bm + rowL + p * 32) * K + kb + colL);
        } else {
            const __bf16* A = (const __bf16*)Aptr;
#pragma unroll
            for (int p = 0; p < 4; p++)
                au2[p] = *(const uint2*)(A + (size_t)(bm + rowL + p * 32) * K + kb + colL);
        }
#pragma unroll
        for (int p = 0; p < 4; p++)
            bu2[p] = *(const uint2*)(Bt + (size_t)(bn + rowL + p * 32) * K + kb + colL);
    };

    auto storeRegs = [&](int buf) {
        if constexpr (A_F32) {
#pragma unroll
            for (int p = 0; p < 4; p++) {
                __bf16* dst = (__bf16*)&Alds[buf][(rowL + p * 32) * LDSW + colL];
                dst[0] = (__bf16)af4[p].x; dst[1] = (__bf16)af4[p].y;
                dst[2] = (__bf16)af4[p].z; dst[3] = (__bf16)af4[p].w;
            }
        } else {
#pragma unroll
            for (int p = 0; p < 4; p++)
                *(uint2*)&Alds[buf][(rowL + p * 32) * LDSW + colL] = au2[p];
        }
#pragma unroll
        for (int p = 0; p < 4; p++)
            *(uint2*)&Blds[buf][(rowL + p * 32) * LDSW + colL] = bu2[p];
    };

    const int nk = K / BK;
    loadRegs(0);
    storeRegs(0);
    __syncthreads();

    for (int p = 0; p < nk; p++) {
        const int buf = p & 1;
        if (p + 1 < nk) loadRegs((p + 1) * BK);  // prefetch next tile to regs

        // ---- WMMA on current buffer ----
#pragma unroll
        for (int mt = 0; mt < 2; mt++) {
            union { uint4 u[2]; v16bfx v; } afr;
            const unsigned short* ap =
                &Alds[buf][(waveM * 32 + mt * 16 + lmod) * LDSW + lhalf * 8];
            afr.u[0] = *(const uint4*)ap;         // K = half*8 .. +7
            afr.u[1] = *(const uint4*)(ap + 16);  // K = 16 + half*8 .. +7
#pragma unroll
            for (int nt = 0; nt < 4; nt++) {
                union { uint4 u[2]; v16bfx v; } bfr;
                const unsigned short* bp =
                    &Blds[buf][(waveN * 64 + nt * 16 + lmod) * LDSW + lhalf * 8];
                bfr.u[0] = *(const uint4*)bp;        // K = half*8 .. +7
                bfr.u[1] = *(const uint4*)(bp + 16); // K = 16 + half*8 .. +7
                acc[mt][nt] = __builtin_amdgcn_wmma_f32_16x16x32_bf16(
                    false, afr.v, false, bfr.v, (short)0, acc[mt][nt],
                    false, false);
            }
        }

        if (p + 1 < nk) storeRegs(buf ^ 1);  // fill other buffer
        __syncthreads();                     // single barrier per K-step
    }

    // ---- epilogue: bias + LeakyReLU, store bf16 (intermediate) or f32 ----
#pragma unroll
    for (int mt = 0; mt < 2; mt++) {
#pragma unroll
        for (int nt = 0; nt < 4; nt++) {
            int gcol = bn + waveN * 64 + nt * 16 + lmod;
            float bv = 0.0f;
            if constexpr (BIAS) bv = bias[gcol];
#pragma unroll
            for (int r = 0; r < 8; r++) {
                int grow = bm + waveM * 32 + mt * 16 + r + lhalf * 8;
                float hv = acc[mt][nt][r] + bv;
                if constexpr (LEAKY) hv = hv >= 0.0f ? hv : LEAKY_SLOPE * hv;
                if constexpr (OUT_BF16)
                    Cbf[(size_t)grow * N + gcol] = (__bf16)hv;
                else
                    Cf[(size_t)grow * N + gcol] = hv;
            }
        }
    }
}

// ---------------------------------------------------------------------------
// Sparsemax + renormalize: one 512-thread block per row of logits[P=512].
// Writes a (f32, output) and a_bf16 (workspace, feeds final GEMM).
// ---------------------------------------------------------------------------
__global__ __launch_bounds__(PP) void sparsemax_kernel(
    const float* __restrict__ logits,
    float* __restrict__ a_out, __bf16* __restrict__ a_bf) {

    __shared__ float sb[PP];   // sorted (descending)
    __shared__ float cb[PP];   // cumsum of sorted
    __shared__ float red[PP];  // reductions

    const int row = blockIdx.x;
    const int t = threadIdx.x;

    float v = logits[(size_t)row * PP + t];
    red[t] = v;
    __syncthreads();
    for (int s = PP / 2; s > 0; s >>= 1) {          // max reduce
        if (t < s) red[t] = fmaxf(red[t], red[t + s]);
        __syncthreads();
    }
    float z = v - red[0];
    __syncthreads();
    sb[t] = z;
    __syncthreads();

    // bitonic sort, descending
    for (int ksz = 2; ksz <= PP; ksz <<= 1) {
        for (int j = ksz >> 1; j > 0; j >>= 1) {
            int ixj = t ^ j;
            if (ixj > t) {
                float av = sb[t], bv2 = sb[ixj];
                bool up = ((t & ksz) == 0);
                if (up ? (av < bv2) : (av > bv2)) { sb[t] = bv2; sb[ixj] = av; }
            }
            __syncthreads();
        }
    }

    // inclusive scan of sorted values
    cb[t] = sb[t];
    __syncthreads();
    for (int off = 1; off < PP; off <<= 1) {
        float add = (t >= off) ? cb[t - off] : 0.0f;
        float nv = cb[t] + add;
        __syncthreads();
        cb[t] = nv;
        __syncthreads();
    }

    // support size
    float flag = (1.0f + (float)(t + 1) * sb[t] > cb[t]) ? 1.0f : 0.0f;
    red[t] = flag;
    __syncthreads();
    for (int s = PP / 2; s > 0; s >>= 1) {
        if (t < s) red[t] += red[t + s];
        __syncthreads();
    }
    int kk = (int)red[0];
    if (kk < 1) kk = 1;
    float tau = (cb[kk - 1] - 1.0f) / (float)kk;
    __syncthreads();

    // p = clip(z - tau, 0); renormalize by actual sum
    float pcl = fmaxf(z - tau, 0.0f);
    red[t] = pcl;
    __syncthreads();
    for (int s = PP / 2; s > 0; s >>= 1) {
        if (t < s) red[t] += red[t + s];
        __syncthreads();
    }
    float a = pcl / fmaxf(red[0], 1e-8f);

    a_out[(size_t)row * PP + t] = a;
    a_bf[(size_t)row * PP + t] = (__bf16)a;
}

// ---------------------------------------------------------------------------
// Host-side launcher
// ---------------------------------------------------------------------------
extern "C" void kernel_launch(void* const* d_in, const int* in_sizes, int n_in,
                              void* d_out, int out_size, void* d_ws, size_t ws_size,
                              hipStream_t stream) {
    (void)in_sizes; (void)n_in; (void)out_size; (void)ws_size;

    const float* x     = (const float*)d_in[0];
    const float* W1    = (const float*)d_in[1];
    const float* b1    = (const float*)d_in[2];
    const float* g1    = (const float*)d_in[3];
    const float* be1   = (const float*)d_in[4];
    const float* m1    = (const float*)d_in[5];
    const float* v1    = (const float*)d_in[6];
    const float* W2    = (const float*)d_in[7];
    const float* b2    = (const float*)d_in[8];
    const float* g2    = (const float*)d_in[9];
    const float* be2   = (const float*)d_in[10];
    const float* m2    = (const float*)d_in[11];
    const float* v2    = (const float*)d_in[12];
    const float* Wl    = (const float*)d_in[13];
    const float* bl    = (const float*)d_in[14];
    const float* basis = (const float*)d_in[15];

    // Output layout: a [B*P] | z_bio [B*L] | logits [B*P]
    float* a_out      = (float*)d_out;
    float* z_out      = a_out + (size_t)BQ * PP;
    float* logits_out = z_out + (size_t)BQ * LL;

    // Workspace layout (all bf16 weights stored TRANSPOSED: [N][K])
    char* ws = (char*)d_ws;
    __bf16* W1t    = (__bf16*)ws;                        ws += (size_t)DD  * HH1 * 2;
    __bf16* W2t    = (__bf16*)ws;                        ws += (size_t)HH1 * HH2 * 2;
    __bf16* Wlt    = (__bf16*)ws;                        ws += (size_t)HH2 * PP  * 2;
    __bf16* nbt    = (__bf16*)ws;                        ws += (size_t)PP  * LL  * 2;
    float*  bias1f = (float*)ws;                         ws += (size_t)HH1 * 4;
    float*  bias2f = (float*)ws;                         ws += (size_t)HH2 * 4;
    __bf16* h1bf   = (__bf16*)ws;                        ws += (size_t)BQ * HH1 * 2;
    __bf16* h2bf   = (__bf16*)ws;                        ws += (size_t)BQ * HH2 * 2;
    __bf16* abf    = (__bf16*)ws;                        ws += (size_t)BQ * PP  * 2;

    // ---- prep: fold BN into weights, transpose to [N][K] bf16
    fold_transpose_kernel<true>
        <<<dim3(HH1 / 64, DD / 64), 256, 0, stream>>>(W1, g1, v1, W1t, DD, HH1);
    fold_transpose_kernel<true>
        <<<dim3(HH2 / 64, HH1 / 64), 256, 0, stream>>>(W2, g2, v2, W2t, HH1, HH2);
    fold_transpose_kernel<false>
        <<<dim3(PP / 64, HH2 / 64), 256, 0, stream>>>(Wl, nullptr, nullptr, Wlt, HH2, PP);
    fold_bias_kernel<<<HH1 / 256, 256, 0, stream>>>(b1, g1, be1, m1, v1, bias1f, HH1);
    fold_bias_kernel<<<HH2 / 256, 256, 0, stream>>>(b2, g2, be2, m2, v2, bias2f, HH2);
    normalize_basis_kernel<<<PP, LL, 0, stream>>>(basis, nbt);

    // ---- GEMM 1: h1 = leaky(x @ W1' + bias1')  [f32 A -> bf16 out]
    gemm_wmma_kernel<true, true, true, true>
        <<<dim3(HH1 / BN, BQ / BM), 256, 0, stream>>>(
            x, W1t, bias1f, nullptr, h1bf, BQ, HH1, DD);

    // ---- GEMM 2: h2 = leaky(h1 @ W2' + bias2')
    gemm_wmma_kernel<false, true, true, true>
        <<<dim3(HH2 / BN, BQ / BM), 256, 0, stream>>>(
            h1bf, W2t, bias2f, nullptr, h2bf, BQ, HH2, HH1);

    // ---- GEMM 3: logits = h2 @ Wl + bl   [f32 out, an output tensor]
    gemm_wmma_kernel<false, false, true, false>
        <<<dim3(PP / BN, BQ / BM), 256, 0, stream>>>(
            h2bf, Wlt, bl, logits_out, nullptr, BQ, PP, HH2);

    // ---- sparsemax + renormalize  -> a (f32 out) + a_bf16 (ws)
    sparsemax_kernel<<<BQ, PP, 0, stream>>>(logits_out, a_out, abf);

    // ---- GEMM 4: z_bio = a @ nb   [f32 out]
    gemm_wmma_kernel<false, false, false, false>
        <<<dim3(LL / BN, BQ / BM), 256, 0, stream>>>(
            abf, nbt, nullptr, z_out, nullptr, BQ, LL, PP);
}
